// SAGE_87084756893761
// MI455X (gfx1250) — matrix operand order repaired and checked by reading, not compiled, not run on hardware
//
#include <hip/hip_runtime.h>
#include <hip/hip_bf16.h>

// ---------------------------------------------------------------------------
// GraphSAGE 2-layer forward for MI455X (gfx1250), wave32 + WMMA bf16 split.
//   layer: agg = scatter_sum(x[src] -> dst) * inv_deg
//          y   = agg @ Wl^T + bl + x @ Wr^T        (ReLU after layer 0)
//   out = log_softmax(y1)
// GEMMs: v_wmma_f32_16x16x32_bf16 with hi/lo split (~fp32 accuracy).
// Weights pre-split to bf16 hi/lo once per launch; A split hoisted per tile.
// ---------------------------------------------------------------------------

#define DIN  128
#define DH   128
#define DOUT 40
#define DOUTP 48   // DOUT padded to 16

typedef __attribute__((ext_vector_type(16))) __bf16 v16bf;
typedef __attribute__((ext_vector_type(8)))  float  v8f;

__device__ inline v8f wmma_bf16(v16bf a, v16bf b, v8f c) {
  // (neg_a, A, neg_b, B, c_mod, C, reuse_a, reuse_b)
  return __builtin_amdgcn_wmma_f32_16x16x32_bf16(false, a, false, b,
                                                 (short)0, c, false, false);
}

// A-fragment (16x32 bf16, M x K): lane group g=0 holds K {0..15},
// g=1 holds K {16..31} of row M = lane&15.  rowptr -> row base + kc*32.
__device__ inline void load_frag_a(const float* __restrict__ rowptr, float scale,
                                   int g, v16bf& hi, v16bf& lo) {
  const float4* p0 = reinterpret_cast<const float4*>(rowptr + g * 16);
  const float4* p1 = reinterpret_cast<const float4*>(rowptr + g * 16 + 8);
  float4 t0 = p0[0], t1 = p0[1], t2 = p1[0], t3 = p1[1];
  float v[16] = {t0.x, t0.y, t0.z, t0.w, t1.x, t1.y, t1.z, t1.w,
                 t2.x, t2.y, t2.z, t2.w, t3.x, t3.y, t3.z, t3.w};
#pragma unroll
  for (int i = 0; i < 16; ++i) {
    float f = v[i] * scale;
    __bf16 h = (__bf16)f;
    hi[i] = h;
    lo[i] = (__bf16)(f - (float)h);
  }
}

// ---------------------------------------------------------------------------
// Weight pre-split: f32 [rows][128] -> bf16 hi/lo [rowsPad][128] (pad = 0).
// ---------------------------------------------------------------------------
__global__ __launch_bounds__(256) void split_w_kernel(
    const float* __restrict__ w, __bf16* __restrict__ hi,
    __bf16* __restrict__ lo, int rows, int rowsPad) {
  int i = blockIdx.x * blockDim.x + threadIdx.x;
  int total = rowsPad * 128;
  if (i >= total) return;
  int r = i >> 7;
  float f = (r < rows) ? w[i] : 0.0f;
  __bf16 h = (__bf16)f;
  hi[i] = h;
  lo[i] = (__bf16)(f - (float)h);
}

// ---------------------------------------------------------------------------
// Degree / scatter kernels (L2-resident atomics)
// ---------------------------------------------------------------------------
__global__ __launch_bounds__(256) void deg_kernel(const int* __restrict__ dst,
                                                  float* __restrict__ deg, int E) {
  int e = blockIdx.x * blockDim.x + threadIdx.x;
  if (e < E) atomicAdd(&deg[dst[e]], 1.0f);
}

__global__ __launch_bounds__(256) void invdeg_kernel(float* __restrict__ deg, int N) {
  int i = blockIdx.x * blockDim.x + threadIdx.x;
  if (i < N) deg[i] = 1.0f / fmaxf(deg[i], 1.0f);
}

// one thread per (edge, feature): 128 consecutive threads share one edge ->
// coalesced row read + coalesced atomic row update (agg fits in 192MB L2).
__global__ __launch_bounds__(256) void scatter_kernel(
    const float* __restrict__ feat, const int* __restrict__ src,
    const int* __restrict__ dst, float* __restrict__ agg, long long total) {
  long long idx = (long long)blockIdx.x * blockDim.x + threadIdx.x;
  if (idx >= total) return;
  int e = (int)(idx >> 7);
  int f = (int)(idx & 127);
  int s = src[e], d = dst[e];
  atomicAdd(&agg[(size_t)d * 128 + f], feat[(size_t)s * 128 + f]);
}

// ---------------------------------------------------------------------------
// Fused dual-GEMM:  out = act( (agg*inv_deg) @ Wl^T + bl + feat @ Wr^T )
// One wave per 16-row tile. A split hoisted (8 frags in regs); B loaded
// pre-split (32B contiguous per lane). 24 wmma per 16x16 output tile,
// two independent accumulator chains for ILP.
// ---------------------------------------------------------------------------
template <int FOUT, int FOUTP, bool RELU>
__global__ __launch_bounds__(256) void sage_gemm_kernel(
    const float* __restrict__ agg, const float* __restrict__ feat,
    const float* __restrict__ inv_deg,
    const __bf16* __restrict__ Wlh, const __bf16* __restrict__ Wll,
    const __bf16* __restrict__ Wrh, const __bf16* __restrict__ Wrl,
    const float* __restrict__ bl, float* __restrict__ out,
    int nRowTiles, int N) {
  const int waveId = threadIdx.x >> 5;
  const int lane   = threadIdx.x & 31;
  const int rt     = blockIdx.x * 8 + waveId;
  if (rt >= nRowTiles) return;          // wave-uniform: EXEC all-ones below

  const int mIdx = lane & 15;
  const int g    = lane >> 4;
  int row = rt * 16 + mIdx;
  if (row >= N) row = N - 1;            // safe clamp for loads (stores guarded)
  const float scale = inv_deg[row];
  const float* aggrow = agg  + (size_t)row * 128;
  const float* xrow   = feat + (size_t)row * 128;

  // Hoisted A fragments: split once per row tile, reused for all N tiles.
  v16bf aggH[4], aggL[4], xH[4], xL[4];
#pragma unroll
  for (int kc = 0; kc < 4; ++kc) {
    load_frag_a(aggrow + kc * 32, scale, g, aggH[kc], aggL[kc]);
    load_frag_a(xrow   + kc * 32, 1.0f,  g, xH[kc],  xL[kc]);
  }

  const int NT = FOUTP / 16;
  for (int nt = 0; nt < NT; ++nt) {
    v8f accL = {0.f, 0.f, 0.f, 0.f, 0.f, 0.f, 0.f, 0.f};
    v8f accR = {0.f, 0.f, 0.f, 0.f, 0.f, 0.f, 0.f, 0.f};
    const int wrow = nt * 16 + mIdx;     // output column owned by lane (< FOUTP)
    const size_t wBase = (size_t)wrow * 128 + g * 16;

#pragma unroll
    for (int kc = 0; kc < 4; ++kc) {
      const size_t o = wBase + kc * 32;
      v16bf blh = *reinterpret_cast<const v16bf*>(Wlh + o);
      v16bf bll = *reinterpret_cast<const v16bf*>(Wll + o);
      v16bf brh = *reinterpret_cast<const v16bf*>(Wrh + o);
      v16bf brl = *reinterpret_cast<const v16bf*>(Wrl + o);
      // (Ah+Al)(Bh+Bl) ~= AhBh + AhBl + AlBh   (f32 accumulate)
      accL = wmma_bf16(aggH[kc], blh, accL);
      accR = wmma_bf16(xH[kc],   brh, accR);
      accL = wmma_bf16(aggH[kc], bll, accL);
      accR = wmma_bf16(xH[kc],   brl, accR);
      accL = wmma_bf16(aggL[kc], blh, accL);
      accR = wmma_bf16(xL[kc],   brh, accR);
    }

    const bool wvalid = (wrow < FOUT);
    const float bias = wvalid ? bl[wrow] : 0.0f;
#pragma unroll
    for (int v = 0; v < 8; ++v) {
      const int mr   = g * 8 + v;        // D row for acc element v
      const int orow = rt * 16 + mr;
      float val = accL[v] + accR[v] + bias;
      if (RELU) val = fmaxf(val, 0.0f);
      if (wvalid && orow < N) out[(size_t)orow * FOUT + wrow] = val;
    }
  }
}

// ---------------------------------------------------------------------------
// log_softmax over 40 classes, one thread per node
// ---------------------------------------------------------------------------
__global__ __launch_bounds__(256) void logsoftmax_kernel(
    const float* __restrict__ in, float* __restrict__ out, int N) {
  int i = blockIdx.x * blockDim.x + threadIdx.x;
  if (i >= N) return;
  const float* r = in + (size_t)i * DOUT;
  float m = -3.402823466e+38f;
#pragma unroll
  for (int j = 0; j < DOUT; ++j) m = fmaxf(m, r[j]);
  float s = 0.0f;
#pragma unroll
  for (int j = 0; j < DOUT; ++j) s += expf(r[j] - m);
  float ls = logf(s);
  float* o = out + (size_t)i * DOUT;
#pragma unroll
  for (int j = 0; j < DOUT; ++j) o[j] = r[j] - m - ls;
}

// ---------------------------------------------------------------------------
extern "C" void kernel_launch(void* const* d_in, const int* in_sizes, int n_in,
                              void* d_out, int out_size, void* d_ws, size_t ws_size,
                              hipStream_t stream) {
  (void)n_in; (void)out_size; (void)ws_size;
  const float* x   = (const float*)d_in[0];
  const int*   ei  = (const int*)d_in[1];
  const float* Wl0 = (const float*)d_in[2];
  const float* bl0 = (const float*)d_in[3];
  const float* Wr0 = (const float*)d_in[4];
  const float* Wl1 = (const float*)d_in[5];
  const float* bl1 = (const float*)d_in[6];
  const float* Wr1 = (const float*)d_in[7];

  const int N = in_sizes[0] / DIN;
  const int E = in_sizes[1] / 2;
  const int* src = ei;
  const int* dst = ei + E;

  // workspace carve: f32 region then 256B-aligned bf16 weight-split region
  float* inv_deg = (float*)d_ws;
  float* aggf    = inv_deg + N;
  float* h       = aggf + (size_t)N * DIN;
  float* logits  = h + (size_t)N * DH;
  uintptr_t bfBase = (uintptr_t)(logits + (size_t)N * DOUT);
  bfBase = (bfBase + 255) & ~(uintptr_t)255;
  __bf16* w0lh = (__bf16*)bfBase;
  __bf16* w0ll = w0lh + DH * 128;
  __bf16* w0rh = w0ll + DH * 128;
  __bf16* w0rl = w0rh + DH * 128;
  __bf16* w1lh = w0rl + DH * 128;
  __bf16* w1ll = w1lh + DOUTP * 128;
  __bf16* w1rh = w1ll + DOUTP * 128;
  __bf16* w1rl = w1rh + DOUTP * 128;

  // ---- prep: degree + weight splits (tiny) ----
  hipMemsetAsync(inv_deg, 0, sizeof(float) * (size_t)N, stream);
  hipMemsetAsync(aggf, 0, sizeof(float) * (size_t)N * DIN, stream);

  split_w_kernel<<<(DH * 128 + 255) / 256, 256, 0, stream>>>(Wl0, w0lh, w0ll, DH, DH);
  split_w_kernel<<<(DH * 128 + 255) / 256, 256, 0, stream>>>(Wr0, w0rh, w0rl, DH, DH);
  split_w_kernel<<<(DOUTP * 128 + 255) / 256, 256, 0, stream>>>(Wl1, w1lh, w1ll, DOUT, DOUTP);
  split_w_kernel<<<(DOUTP * 128 + 255) / 256, 256, 0, stream>>>(Wr1, w1rh, w1rl, DOUT, DOUTP);

  deg_kernel<<<(E + 255) / 256, 256, 0, stream>>>(dst, inv_deg, E);
  invdeg_kernel<<<(N + 255) / 256, 256, 0, stream>>>(inv_deg, N);

  // ---- layer 0 ----
  const long long tot = (long long)E * 128;
  const unsigned scatBlocks = (unsigned)((tot + 255) / 256);
  scatter_kernel<<<scatBlocks, 256, 0, stream>>>(x, src, dst, aggf, tot);

  const int nRT = (N + 15) / 16;
  const int gemmBlocks = (nRT + 7) / 8;
  sage_gemm_kernel<DH, DH, true><<<gemmBlocks, 256, 0, stream>>>(
      aggf, x, inv_deg, w0lh, w0ll, w0rh, w0rl, bl0, h, nRT, N);

  // ---- layer 1 ----
  hipMemsetAsync(aggf, 0, sizeof(float) * (size_t)N * DH, stream);
  scatter_kernel<<<scatBlocks, 256, 0, stream>>>(h, src, dst, aggf, tot);
  sage_gemm_kernel<DOUT, DOUTP, false><<<gemmBlocks, 256, 0, stream>>>(
      aggf, h, inv_deg, w1lh, w1ll, w1rh, w1rl, bl1, logits, nRT, N);

  logsoftmax_kernel<<<(N + 255) / 256, 256, 0, stream>>>(logits, (float*)d_out, N);
}
